// Model_28132035789456
// MI455X (gfx1250) — matrix operand order repaired and checked by reading, not compiled, not run on hardware
//
#include <hip/hip_runtime.h>

// ---------------------------------------------------------------------------
// Types for CDNA5 WMMA (wave32, v_wmma_f32_16x16x32_f16)
// ---------------------------------------------------------------------------
typedef __attribute__((ext_vector_type(16))) _Float16 v16h;
typedef __attribute__((ext_vector_type(8)))  _Float16 v8h;
typedef __attribute__((ext_vector_type(8)))  float    v8f;

#define TILE_M 128
#define TILE_N 128
#define TILE_K 32
#define LDSTR  40            // 32 + 8 element pad (80B rows, 16B aligned)
#define BUF_ELEMS ((TILE_M + TILE_N) * LDSTR)   // one double-buffer slice

// Epilogue modes
#define EPI_NONE      0
#define EPI_GATE_RES  1   // out = g*(acc+b) + g
#define EPI_SOFTPLUS  2   // out = softplus(acc+b)
#define EPI_RES_SILU  3   // out = resid + silu(acc+b)

__device__ __forceinline__ float siluf(float v) {
    return v / (1.0f + __expf(-v));
}

// gfx1250 async direct global->LDS copy (bypasses VGPRs, tracked by ASYNCcnt)
__device__ __forceinline__ void async_ld_b128(unsigned int lds_off, const void* gptr) {
    asm volatile("global_load_async_to_lds_b128 %0, %1, off"
                 :: "v"(lds_off), "v"(gptr) : "memory");
}
__device__ __forceinline__ void wait_async4() {
    asm volatile("s_wait_asynccnt 0x4" ::: "memory");
}
__device__ __forceinline__ void wait_async0() {
    asm volatile("s_wait_asynccnt 0x0" ::: "memory");
}

// ---------------------------------------------------------------------------
// Generic tiled WMMA GEMM: C[M,N] = epi(A[M,K] @ W[N,K]^T + bias)
// A, W in f16; f32 accumulate; double-buffered async global->LDS staging.
// Block 128x128, 8 waves (2 along M x 4 along N), wave tile 64x32.
// ---------------------------------------------------------------------------
__global__ __launch_bounds__(256)
void gemm_wmma_f16(const _Float16* __restrict__ A, int lda,
                   const _Float16* __restrict__ W, int ldb,
                   const float* __restrict__ bias,
                   const float* __restrict__ gate,
                   const float* __restrict__ resid,
                   float* __restrict__ Cf, _Float16* __restrict__ Ch, int ldc,
                   int Mdim, int Ndim, int Kdim, int mode)
{
    __shared__ _Float16 lds[2 * BUF_ELEMS];   // [buf][A 128xLDSTR | B 128xLDSTR]

    const int tid  = threadIdx.x;
    const int lane = tid & 31;
    const int wave = tid >> 5;
    const int wm   = wave & 1;    // 2 waves along M (64 rows each)
    const int wn   = wave >> 1;   // 4 waves along N (32 cols each)
    const int half = lane >> 4;
    const int l16  = lane & 15;

    const int m0 = blockIdx.y * TILE_M;
    const int n0 = blockIdx.x * TILE_N;

    // ---- per-thread staging coordinates (2 x b128 for A, 2 x b128 for B) ----
    const int srow = tid >> 1;          // 0..127 tile row
    const int selt = (tid & 1) * 16;    // element offset within row: 0 or 16
    // clamp OOB rows instead of zero-fill: garbage only feeds columns >= Ndim,
    // which the epilogue never stores.
    int gmA = m0 + srow; if (gmA > Mdim - 1) gmA = Mdim - 1;
    int gnB = n0 + srow; if (gnB > Ndim - 1) gnB = Ndim - 1;
    const _Float16* gA = A + (size_t)gmA * lda + selt;
    const _Float16* gB = W + (size_t)gnB * ldb + selt;

    const unsigned int ldsBase = (unsigned int)(size_t)(&lds[0]);
    const unsigned int aOff    = ldsBase + (unsigned int)((srow * LDSTR + selt) * 2);
    const unsigned int bOff    = aOff + (unsigned int)(TILE_M * LDSTR * 2);

    v8f acc[4][2];
#pragma unroll
    for (int mt = 0; mt < 4; ++mt)
#pragma unroll
        for (int nt = 0; nt < 2; ++nt)
            acc[mt][nt] = (v8f){0.f,0.f,0.f,0.f,0.f,0.f,0.f,0.f};

    const int kTiles = Kdim / TILE_K;

    // prologue: stage k-tile 0 into buffer 0
    async_ld_b128(aOff,      gA);
    async_ld_b128(aOff + 16, gA + 8);
    async_ld_b128(bOff,      gB);
    async_ld_b128(bOff + 16, gB + 8);

    for (int kt = 0; kt < kTiles; ++kt) {
        const int cur = kt & 1;
        if (kt + 1 < kTiles) {
            // prefetch next k-tile into the other buffer while we compute
            const unsigned int bufo = (unsigned int)(((cur ^ 1) * BUF_ELEMS) * 2);
            const int kb = (kt + 1) * TILE_K;
            async_ld_b128(aOff + bufo,      gA + kb);
            async_ld_b128(aOff + bufo + 16, gA + kb + 8);
            async_ld_b128(bOff + bufo,      gB + kb);
            async_ld_b128(bOff + bufo + 16, gB + kb + 8);
            wait_async4();   // current tile's 4 copies done; next tile in flight
        } else {
            wait_async0();
        }
        __syncthreads();

        const _Float16* ldsAb = &lds[cur * BUF_ELEMS];
        const _Float16* ldsBb = ldsAb + TILE_M * LDSTR;

        // ---- B fragments: lane=(half,col), K run of 16 at half*16 ----
        v16h bfrag[2];
#pragma unroll
        for (int nt = 0; nt < 2; ++nt) {
            const _Float16* p = &ldsBb[(wn * 32 + nt * 16 + l16) * LDSTR + half * 16];
            v8h lo = *(const v8h*)(p);
            v8h hi = *(const v8h*)(p + 8);
            bfrag[nt] = __builtin_shufflevector(lo, hi,
                0,1,2,3,4,5,6,7,8,9,10,11,12,13,14,15);
        }
        // ---- A fragments + 8 WMMAs per K-step ----
#pragma unroll
        for (int mt = 0; mt < 4; ++mt) {
            const _Float16* p = &ldsAb[(wm * 64 + mt * 16 + l16) * LDSTR];
            v8h lo = *(const v8h*)(p + half * 16);
            v8h hi = *(const v8h*)(p + half * 16 + 8);
            v16h afrag = __builtin_shufflevector(lo, hi,
                0,1,2,3,4,5,6,7,8,9,10,11,12,13,14,15);
#pragma unroll
            for (int nt = 0; nt < 2; ++nt) {
                acc[mt][nt] = __builtin_amdgcn_wmma_f32_16x16x32_f16(
                    false, afrag, false, bfrag[nt],
                    (short)0, acc[mt][nt], false, false);
            }
        }
        __syncthreads();   // all waves done reading buf[cur] before it is re-filled
    }

    // ---- epilogue: C layout row = half*8 + j, col = lane&15 ----
#pragma unroll
    for (int mt = 0; mt < 4; ++mt) {
#pragma unroll
        for (int nt = 0; nt < 2; ++nt) {
            int col = n0 + wn * 32 + nt * 16 + l16;
            if (col >= Ndim) continue;
            float bv = bias ? bias[col] : 0.0f;
#pragma unroll
            for (int j = 0; j < 8; ++j) {
                int row = m0 + wm * 64 + mt * 16 + half * 8 + j;
                if (row >= Mdim) continue;
                size_t idx = (size_t)row * ldc + col;
                float v = acc[mt][nt][j] + bv;
                if (mode == EPI_GATE_RES) {
                    float g = gate[idx];
                    v = g * v + g;
                } else if (mode == EPI_SOFTPLUS) {
                    v = (v > 20.0f) ? v : log1pf(__expf(v));
                } else if (mode == EPI_RES_SILU) {
                    v = resid[idx] + siluf(v);
                }
                if (Cf) Cf[idx] = v;
                if (Ch) Ch[idx] = (_Float16)v;
            }
        }
    }
}

// ---------------------------------------------------------------------------
// f32 -> f16 conversion (weights)
// ---------------------------------------------------------------------------
__global__ void cvt_f32_f16(const float* __restrict__ s, _Float16* __restrict__ d, long n)
{
    for (long i = (long)blockIdx.x * blockDim.x + threadIdx.x; i < n;
         i += (long)gridDim.x * blockDim.x)
        d[i] = (_Float16)s[i];
}

// ---------------------------------------------------------------------------
// Embedding gather + concat -> x0 (f32 for gate epilogue, f16 for GEMM)
// ---------------------------------------------------------------------------
__global__ void gather_concat(const float* __restrict__ hs,
                              const int* __restrict__ ids,
                              const float* __restrict__ emb,
                              float* __restrict__ x0f, _Float16* __restrict__ x0h,
                              int Mrows, int Hdim, int H2dim)
{
    long total = (long)Mrows * H2dim;
    for (long i = (long)blockIdx.x * blockDim.x + threadIdx.x; i < total;
         i += (long)gridDim.x * blockDim.x) {
        int m = (int)(i / H2dim);
        int c = (int)(i % H2dim);
        float v = (c < Hdim) ? emb[(size_t)ids[m] * Hdim + c]
                             : hs[(size_t)m * Hdim + (c - Hdim)];
        x0f[i] = v;
        x0h[i] = (_Float16)v;
    }
}

// ---------------------------------------------------------------------------
// Depthwise causal conv (K=4) over time + SiLU, reads x half of xz
// ---------------------------------------------------------------------------
__global__ void conv_silu(const float* __restrict__ xz,
                          const float* __restrict__ cw, const float* __restrict__ cb,
                          float* __restrict__ uf, _Float16* __restrict__ uh,
                          int Bb, int Ll, int Edim, int E2dim, int Kk)
{
    long total = (long)Bb * Ll * Edim;
    for (long i = (long)blockIdx.x * blockDim.x + threadIdx.x; i < total;
         i += (long)gridDim.x * blockDim.x) {
        int e  = (int)(i % Edim);
        long bl = i / Edim;
        int l  = (int)(bl % Ll);
        int b  = (int)(bl / Ll);
        float acc = cb[e];
#pragma unroll
        for (int k = 0; k < 4; ++k) {
            int ls = l - (Kk - 1) + k;
            if (ls >= 0)
                acc += xz[((size_t)b * Ll + ls) * E2dim + e] * cw[e * Kk + k];
        }
        float u = siluf(acc);
        uf[i] = u;
        uh[i] = (_Float16)u;
    }
}

// ---------------------------------------------------------------------------
// Selective scan: one thread per (b,e) channel, h[16] in regs, B/C via LDS.
// Fuses  y = (scan + u*D) * silu(z)  into the store (f16 for next GEMM).
// ---------------------------------------------------------------------------
__global__ __launch_bounds__(256)
void ssm_scan(const float* __restrict__ delta, const float* __restrict__ dbc,
              const float* __restrict__ u, const float* __restrict__ xz,
              const float* __restrict__ A_log, const float* __restrict__ Dp,
              _Float16* __restrict__ yh,
              int Bb, int Ll, int Edim, int E2dim, int Rr, int Nn, int dbcStride)
{
    __shared__ float sB[16];
    __shared__ float sC[16];
    const int e = blockIdx.x * blockDim.x + threadIdx.x;
    const int b = blockIdx.y;

    float Arow[16], h[16];
#pragma unroll
    for (int n = 0; n < 16; ++n) {
        Arow[n] = -__expf(A_log[(size_t)e * Nn + n]);
        h[n] = 0.0f;
    }
    const float De = Dp[e];

    for (int l = 0; l < Ll; ++l) {
        size_t rowBL = (size_t)b * Ll + l;
        if (threadIdx.x < 32) {
            float v = dbc[rowBL * dbcStride + Rr + threadIdx.x];
            if (threadIdx.x < 16) sB[threadIdx.x] = v;
            else                  sC[threadIdx.x - 16] = v;
        }
        __syncthreads();
        float d  = delta[rowBL * Edim + e];
        float ut = u[rowBL * Edim + e];
        float du = d * ut;
        float y  = 0.0f;
#pragma unroll
        for (int n = 0; n < 16; ++n) {
            h[n] = __expf(d * Arow[n]) * h[n] + du * sB[n];
            y += h[n] * sC[n];
        }
        float z = xz[rowBL * E2dim + Edim + e];
        float out = (y + ut * De) * siluf(z);
        yh[rowBL * Edim + e] = (_Float16)out;
        __syncthreads();
    }
}

// ---------------------------------------------------------------------------
// Host-side orchestration
// ---------------------------------------------------------------------------
extern "C" void kernel_launch(void* const* d_in, const int* in_sizes, int n_in,
                              void* d_out, int out_size, void* d_ws, size_t ws_size,
                              hipStream_t stream)
{
    (void)in_sizes; (void)n_in; (void)out_size; (void)ws_size;

    const int Bb = 2, Ll = 1024, Hh = 2048, Ee = 4096, Nn = 16, Kk = 4, Rr = 128;
    const int M   = Bb * Ll;      // 2048 rows
    const int H2  = 2 * Hh;       // 4096
    const int E2  = 2 * Ee;       // 8192
    const int DBC = Rr + 2 * Nn;  // 160

    const float* hidden  = (const float*)d_in[0];
    const int*   ids     = (const int*)  d_in[1];
    const float* emb     = (const float*)d_in[2];
    const float* W_c     = (const float*)d_in[3];
    const float* b_c     = (const float*)d_in[4];
    const float* W_fc    = (const float*)d_in[5];
    const float* b_fc    = (const float*)d_in[6];
    const float* W_in    = (const float*)d_in[7];
    const float* conv_w  = (const float*)d_in[8];
    const float* conv_b  = (const float*)d_in[9];
    const float* W_xproj = (const float*)d_in[10];
    const float* W_dt    = (const float*)d_in[11];
    const float* b_dt    = (const float*)d_in[12];
    const float* A_log   = (const float*)d_in[13];
    const float* Dv      = (const float*)d_in[14];
    const float* W_out   = (const float*)d_in[15];
    const float* W_res   = (const float*)d_in[16];
    const float* b_res   = (const float*)d_in[17];

    char* ws = (char*)d_ws;
    size_t off = 0;
    auto alloc = [&](size_t bytes) -> char* {
        char* p = ws + off;
        off = (off + bytes + 255) & ~(size_t)255;
        return p;
    };

    float*     x0f     = (float*)    alloc((size_t)M * H2 * 4);
    _Float16*  x0h     = (_Float16*) alloc((size_t)M * H2 * 2);
    _Float16*  Wc16    = (_Float16*) alloc((size_t)H2 * H2 * 2);
    _Float16*  xc16    = (_Float16*) alloc((size_t)M * H2 * 2);
    _Float16*  Wfc16   = (_Float16*) alloc((size_t)Hh * H2 * 2);
    _Float16*  hs16    = (_Float16*) alloc((size_t)M * Hh * 2);
    _Float16*  Win16   = (_Float16*) alloc((size_t)E2 * Hh * 2);
    float*     xzf     = (float*)    alloc((size_t)M * E2 * 4);
    float*     uf      = (float*)    alloc((size_t)M * Ee * 4);
    _Float16*  uh      = (_Float16*) alloc((size_t)M * Ee * 2);
    _Float16*  Wxp16   = (_Float16*) alloc((size_t)DBC * Ee * 2);
    float*     dbcf    = (float*)    alloc((size_t)M * DBC * 4);
    _Float16*  dbch    = (_Float16*) alloc((size_t)M * DBC * 2);
    _Float16*  Wdt16   = (_Float16*) alloc((size_t)Ee * Rr * 2);
    float*     deltaf  = (float*)    alloc((size_t)M * Ee * 4);
    _Float16*  yh      = (_Float16*) alloc((size_t)M * Ee * 2);
    _Float16*  Wout16  = (_Float16*) alloc((size_t)Hh * Ee * 2);
    float*     opref   = (float*)    alloc((size_t)M * Hh * 4);
    _Float16*  opreh   = (_Float16*) alloc((size_t)M * Hh * 2);
    _Float16*  Wres16  = (_Float16*) alloc((size_t)Hh * Hh * 2);

    // --- weight conversions to f16 ---
    cvt_f32_f16<<<2048, 256, 0, stream>>>(W_c,     Wc16,  (long)H2 * H2);
    cvt_f32_f16<<<2048, 256, 0, stream>>>(W_fc,    Wfc16, (long)Hh * H2);
    cvt_f32_f16<<<2048, 256, 0, stream>>>(W_in,    Win16, (long)E2 * Hh);
    cvt_f32_f16<<<1024, 256, 0, stream>>>(W_xproj, Wxp16, (long)DBC * Ee);
    cvt_f32_f16<<<1024, 256, 0, stream>>>(W_dt,    Wdt16, (long)Ee * Rr);
    cvt_f32_f16<<<2048, 256, 0, stream>>>(W_out,   Wout16,(long)Hh * Ee);
    cvt_f32_f16<<<2048, 256, 0, stream>>>(W_res,   Wres16,(long)Hh * Hh);

    // --- x0 = concat(embed[ids], hidden) ---
    gather_concat<<<4096, 256, 0, stream>>>(hidden, ids, emb, x0f, x0h, M, Hh, H2);

    dim3 blk(256);
    // GEMM1: xc = x0*(x0@Wc^T + b_c) + x0            [2048 x 4096 x 4096]
    gemm_wmma_f16<<<dim3(H2 / TILE_N, M / TILE_M), blk, 0, stream>>>(
        x0h, H2, Wc16, H2, b_c, x0f, nullptr,
        nullptr, xc16, H2, M, H2, H2, EPI_GATE_RES);

    // GEMM2: hs = xc@Wfc^T + b_fc                    [2048 x 2048 x 4096]
    gemm_wmma_f16<<<dim3(Hh / TILE_N, M / TILE_M), blk, 0, stream>>>(
        xc16, H2, Wfc16, H2, b_fc, nullptr, nullptr,
        nullptr, hs16, Hh, M, Hh, H2, EPI_NONE);

    // GEMM3: xz = hs@Win^T                           [2048 x 8192 x 2048]
    gemm_wmma_f16<<<dim3(E2 / TILE_N, M / TILE_M), blk, 0, stream>>>(
        hs16, Hh, Win16, Hh, nullptr, nullptr, nullptr,
        xzf, nullptr, E2, M, E2, Hh, EPI_NONE);

    // conv + silu -> u
    conv_silu<<<4096, 256, 0, stream>>>(xzf, conv_w, conv_b, uf, uh, Bb, Ll, Ee, E2, Kk);

    // GEMM4: dbc = u@Wxproj^T                        [2048 x 160 x 4096]
    gemm_wmma_f16<<<dim3((DBC + TILE_N - 1) / TILE_N, M / TILE_M), blk, 0, stream>>>(
        uh, Ee, Wxp16, Ee, nullptr, nullptr, nullptr,
        dbcf, dbch, DBC, M, DBC, Ee, EPI_NONE);

    // GEMM5: delta = softplus(dbc[:, :R]@Wdt^T + b_dt)  [2048 x 4096 x 128]
    gemm_wmma_f16<<<dim3(Ee / TILE_N, M / TILE_M), blk, 0, stream>>>(
        dbch, DBC, Wdt16, Rr, b_dt, nullptr, nullptr,
        deltaf, nullptr, Ee, M, Ee, Rr, EPI_SOFTPLUS);

    // selective scan, fused with (+u*D) * silu(z)
    ssm_scan<<<dim3(Ee / 256, Bb), blk, 0, stream>>>(
        deltaf, dbcf, uf, xzf, A_log, Dv, yh, Bb, Ll, Ee, E2, Rr, Nn, DBC);

    // GEMM6: out_pre = y@Wout^T                      [2048 x 2048 x 4096]
    gemm_wmma_f16<<<dim3(Hh / TILE_N, M / TILE_M), blk, 0, stream>>>(
        yh, Ee, Wout16, Ee, nullptr, nullptr, nullptr,
        opref, opreh, Hh, M, Hh, Ee, EPI_NONE);

    // GEMM7: out = out_pre + silu(out_pre@Wres^T + b_res)  [2048 x 2048 x 2048]
    gemm_wmma_f16<<<dim3(Hh / TILE_N, M / TILE_M), blk, 0, stream>>>(
        opreh, Hh, Wres16, Hh, b_res, nullptr, opref,
        (float*)d_out, nullptr, Hh, M, Hh, Hh, EPI_RES_SILU);
}
